// FourierKANLayer_25091198943575
// MI455X (gfx1250) — matrix operand order repaired and checked by reading, not compiled, not run on hardware
//
#include <hip/hip_runtime.h>
#include <hip/hip_bf16.h>

// FourierKAN forward as an on-the-fly-A GEMM using CDNA5 WMMA f16.
//   y[b,j] = sum_{i,k} cos(x[b,i]*(k+1))*Wc[j,i,k] + sin(...)*Ws[j,i,k]
//
// M=2048, N=64, K=256*300*2. Matrix-pipe bound (~40 GFLOP) vs 42MB unique
// bytes (~1.8us at 23.3TB/s) -> v_wmma_f32_16x16x32_f16, f32 accum.
// Trig via rotation recurrence (1 sincos seed per 8-run, 2-FMA rotations).
// Split-K over inputdim (8 splits) -> 256 blocks for occupancy; deterministic
// partial-sum reduction (no float atomics).

#define INPUTDIM 256
#define OUTDIM   64
#define GRIDSIZE 300
#define BATCH    2048
#define KPAD     320           // 10 chunks of 32
#define NCHUNK   10
#define BM       64            // batch rows per block
#define NWAVE    4             // waves per block (128 threads)
#define NSPLIT   8             // split-K over inputdim

typedef __attribute__((ext_vector_type(16))) _Float16 v16h;
typedef __attribute__((ext_vector_type(8)))  float    v8f;

// Dynamic LDS layout:
//   [0, irange*BM*4)      : xt[i_local][r]  transposed x tile
//   [xt_end, +80KB)       : staged f16 B fragments for current i:
//       frag_id = (cs*NCHUNK + c)*4 + t   (cs: cos/sin, c: k-chunk, t: n-tile)
//       byte addr = frag_id*1024 + lane*32 + s*2, lane = 16*hh + n
//       element (lane, s) = W[cs][t*16 + n][i][c*32 + kk], kk = 8hh + s (s<8), 16 + 8hh + s-8 (s>=8)
#define WB_BYTES   (2 * NCHUNK * 4 * 1024)

__global__ void __launch_bounds__(NWAVE * 32)
fourier_kan_wmma(const float* __restrict__ x,
                 const float* __restrict__ W,    // [2][64][256][300]
                 float* __restrict__ tgt,        // out or partial slices
                 int irange)                     // i's per split
{
    extern __shared__ char smem[];
    float* xt = (float*)smem;                            // [irange][64]
    char*  wb = smem + (size_t)irange * BM * 4;          // f16 fragment region

    const int tid  = threadIdx.x;
    const int lane = tid & 31;
    const int wave = tid >> 5;
    const int m0   = blockIdx.x * BM;
    const int ilo  = blockIdx.y * irange;
    const int ihi  = ilo + irange;
    float* myout   = tgt + (size_t)blockIdx.y * BATCH * OUTDIM;

    // ---- load x tile transposed: xt[ii][r] = x[m0+r][ilo+ii] (coalesced in i)
    for (int f = tid; f < irange * BM; f += NWAVE * 32) {
        int r  = f / irange;
        int ii = f - r * irange;
        xt[ii * BM + r] = x[(size_t)(m0 + r) * INPUTDIM + (ilo + ii)];
    }
    __syncthreads();

    v8f acc0 = {}, acc1 = {}, acc2 = {}, acc3 = {};
    v8f* acc[4] = { &acc0, &acc1, &acc2, &acc3 };

    const int rowsel = (lane >= 16) ? 8 : 0;     // A-layout lane K-base
    const int myrow  = wave * 16 + (lane & 15);

    for (int i = ilo; i < ihi; ++i) {
        __syncthreads();   // previous iteration's fragment reads complete

        // ---- stage W[:, :, i, :] into LDS as swizzled f16 B fragments ----
        // item flat = (cs*32 + p)*KPAD + k ; p = j/2 ; k fastest -> coalesced
        for (int f = tid; f < 2 * 32 * KPAD; f += NWAVE * 32) {
            int pcs = f / KPAD;
            int k   = f - pcs * KPAD;
            int p   = pcs & 31;
            int cs  = pcs >> 5;
            float w0 = 0.0f, w1 = 0.0f;
            if (k < GRIDSIZE) {
                size_t base = ((size_t)(cs * OUTDIM + 2 * p) * INPUTDIM + i) * GRIDSIZE + k;
                w0 = W[base];
                w1 = W[base + GRIDSIZE * INPUTDIM];   // j+1 (j-stride = 256*300)
                // pull next i-slice toward L2 (global_prefetch_b8)
                if (((k & 31) == 0) && (i + 1 < ihi))
                    __builtin_prefetch(&W[base + GRIDSIZE], 0, 1);
            }
            int c = k >> 5, L = k & 31, t = p >> 3, v = p & 7;
            int hh = (L >> 3) & 1, s = (L & 7) + ((L >> 4) << 3);
            unsigned addr = ((unsigned)((cs * NCHUNK + c) * 4 + t) << 10)
                          + ((unsigned)s << 1);
            *(_Float16*)(wb + addr + ((unsigned)(hh * 16 + 2 * v) << 5)) = (_Float16)w0;
            *(_Float16*)(wb + addr + ((unsigned)(hh * 16 + 2 * v + 1) << 5)) = (_Float16)w1;
        }
        __syncthreads();

        // ---- per-lane trig state ----
        const float xv = xt[(i - ilo) * BM + myrow];   // stride-1: no conflicts
        const float cx = __cosf(xv);                   // rotation step (angle x)
        const float sx = __sinf(xv);

        #pragma unroll
        for (int c = 0; c < NCHUNK; ++c) {
            // ---- build A fragments (cos & sin) in ISA 16-bit A layout ----
            // lane<16: K = c*32 + {0..7, 16..23}; lane>=16: +8
            v16h ac, as;
            const int kbase = c * 32 + rowsel;
            #pragma unroll
            for (int run = 0; run < 2; ++run) {
                const int k0 = kbase + run * 16;
                float ang = xv * (float)(k0 + 1);      // freq index k0 -> mult k0+1
                float cc = __cosf(ang);
                float ss = __sinf(ang);
                #pragma unroll
                for (int idx = 0; idx < 8; ++idx) {
                    const int kg = k0 + idx;
                    const bool ok = (kg < GRIDSIZE);
                    ac[run * 8 + idx] = (_Float16)(ok ? cc : 0.0f);
                    as[run * 8 + idx] = (_Float16)(ok ? ss : 0.0f);
                    const float nc = cc * cx - ss * sx;   // rotate by x
                    const float ns = ss * cx + cc * sx;
                    cc = nc; ss = ns;
                }
            }

            // ---- 8 WMMAs: 4 n-tiles x {cos, sin} ----
            #pragma unroll
            for (int t = 0; t < 4; ++t) {
                const unsigned fc = (unsigned)(((0 * NCHUNK + c) * 4 + t) << 10) + (unsigned)(lane << 5);
                const unsigned fs = (unsigned)(((1 * NCHUNK + c) * 4 + t) << 10) + (unsigned)(lane << 5);
                v16h bc = *(const v16h*)(wb + fc);   // 2x ds_load_b128
                v16h bs = *(const v16h*)(wb + fs);
                *acc[t] = __builtin_amdgcn_wmma_f32_16x16x32_f16(
                    false, ac, false, bc, (short)0, *acc[t], false, false);
                *acc[t] = __builtin_amdgcn_wmma_f32_16x16x32_f16(
                    false, as, false, bs, (short)0, *acc[t], false, false);
            }
        }
    }

    // ---- store D tiles: lane<16 -> M=r, lane>=16 -> M=r+8 ; N = lane%16 ----
    const int col   = lane & 15;
    const int rowhi = (lane >> 4) << 3;
    #pragma unroll
    for (int t = 0; t < 4; ++t) {
        #pragma unroll
        for (int r = 0; r < 8; ++r) {
            myout[(size_t)(m0 + wave * 16 + rowhi + r) * OUTDIM + t * 16 + col] = (*acc[t])[r];
        }
    }
}

__global__ void __launch_bounds__(256)
reduce_partials(const float* __restrict__ ws, float* __restrict__ out, int nsplit) {
    int idx = blockIdx.x * 256 + threadIdx.x;
    if (idx < BATCH * OUTDIM) {
        float a = 0.0f;
        for (int s = 0; s < nsplit; ++s)
            a += ws[(size_t)s * BATCH * OUTDIM + idx];
        out[idx] = a;
    }
}

extern "C" void kernel_launch(void* const* d_in, const int* in_sizes, int n_in,
                              void* d_out, int out_size, void* d_ws, size_t ws_size,
                              hipStream_t stream) {
    (void)in_sizes; (void)n_in; (void)out_size;
    const float* x = (const float*)d_in[0];                 // [2048][256] f32
    const float* W = (const float*)d_in[1];                 // [2][64][256][300] f32
    float* out = (float*)d_out;                             // [2048][64] f32

    const size_t part_bytes = (size_t)NSPLIT * BATCH * OUTDIM * sizeof(float);
    const bool  use_split   = (d_ws != nullptr) && (ws_size >= part_bytes);
    const int   nsplit      = use_split ? NSPLIT : 1;
    const int   irange      = INPUTDIM / nsplit;
    float* tgt              = use_split ? (float*)d_ws : out;

    const size_t smem = (size_t)irange * BM * 4 + WB_BYTES;   // 88KB (split) / 144KB
    dim3 grid(BATCH / BM, nsplit);       // 32 x 8 = 256 blocks
    dim3 block(NWAVE * 32);              // 128 threads = 4 wave32
    fourier_kan_wmma<<<grid, block, smem, stream>>>(x, W, tgt, irange);

    if (use_split) {
        dim3 rgrid((BATCH * OUTDIM + 255) / 256);
        reduce_partials<<<rgrid, 256, 0, stream>>>((const float*)d_ws, out, nsplit);
    }
}
